// Architecture_40836549050931
// MI455X (gfx1250) — matrix operand order repaired and checked by reading, not compiled, not run on hardware
//
#include <hip/hip_runtime.h>
#include <cstddef>
#include <cstdint>

#define BSZ 8
#define SEQ 1024
#define DIM 512
#define NH  8
#define DK_ 64
#define BH_ (BSZ*NH)
#define MROWS (BSZ*SEQ)

typedef __attribute__((ext_vector_type(16))) __bf16        v16bf;
typedef __attribute__((ext_vector_type(8)))  float         v8f;
typedef __attribute__((ext_vector_type(4)))  float         fv4;
typedef __attribute__((ext_vector_type(4)))  unsigned int  u32x4;

union BF16Frag {
  v16bf          v;      // WMMA operand
  u32x4          q[2];   // two 16B chunks (b128 loads)
  unsigned short u[16];  // element access
};

__device__ __forceinline__ unsigned short f2bf(float f) {
  unsigned int x = __float_as_uint(f);
  x += 0x7FFFu + ((x >> 16) & 1u);          // round-to-nearest-even
  return (unsigned short)(x >> 16);
}
__device__ __forceinline__ unsigned int pack2bf(float lo, float hi) {
  return (unsigned int)f2bf(lo) | ((unsigned int)f2bf(hi) << 16);
}
__device__ __forceinline__ float bf2f(unsigned short u) {
  return __uint_as_float(((unsigned int)u) << 16);
}

__device__ __forceinline__ v8f wmma_bf16(BF16Frag a, BF16Frag b, v8f c) {
  return __builtin_amdgcn_wmma_f32_16x16x32_bf16(false, a.v, false, b.v,
                                                 (short)0, c, false, false);
}

#define V8F_ZERO {0.f,0.f,0.f,0.f,0.f,0.f,0.f,0.f}

// ---------------------------------------------------------------------------
// Kernel 0: fp32 -> bf16 bulk convert (one-shot; halves matrix-operand traffic)
// ---------------------------------------------------------------------------
__global__ __launch_bounds__(256) void cvt_bf16_kernel(
    const float* __restrict__ x, unsigned short* __restrict__ y, int n)
{
  const int idx = (blockIdx.x * 256 + threadIdx.x) * 8;
  if (idx >= n) return;
  fv4 a = *(const fv4*)(x + idx);
  fv4 b = *(const fv4*)(x + idx + 4);
  u32x4 p;
  p[0] = pack2bf(a[0], a[1]);
  p[1] = pack2bf(a[2], a[3]);
  p[2] = pack2bf(b[0], b[1]);
  p[3] = pack2bf(b[2], b[3]);
  *(u32x4*)(y + idx) = p;
}

// ---------------------------------------------------------------------------
// Kernel 1: Y[m,n] = sum_k A[m,k]*W[n,k] + bias[n], A/W bf16, f32 accumulate.
// One wave computes a 32x64 tile: 8 independent accumulator chains,
// 12 b128 loads : 8 WMMAs per K-step, fragments are contiguous 16B chunks.
// mode 0: bf16 head-major        [(b*NH+hd)*SEQ+s]*64+dk   (mean q/k proj)
// mode 1: mode0 raw + sqrt(clip) into Y1                    (cov  q/k proj)
// mode 2: bf16 transposed        [(b*NH+hd)*64+dk]*SEQ+s    (v proj)
// mode 3: fp32 row-major         [m*DIM+n]                  (final output)
// ---------------------------------------------------------------------------
__global__ __launch_bounds__(32) void gemm_bf16_kernel(
    const unsigned short* __restrict__ A, const unsigned short* __restrict__ W,
    const float* __restrict__ bias, void* __restrict__ Y0,
    unsigned short* __restrict__ Y1, int mode)
{
  const int tile = blockIdx.x;            // (MROWS/32) * (DIM/64) tiles
  const int m0 = (tile >> 3) << 5;        // DIM/64 = 8 tiles along N
  const int n0 = (tile & 7) << 6;

  const int lane = threadIdx.x;
  const int l = lane & 15;
  const int h = lane >> 4;

  const unsigned short* ap[2];
  const unsigned short* wp[4];
  #pragma unroll
  for (int mt = 0; mt < 2; ++mt)
    ap[mt] = A + (size_t)(m0 + (mt << 4) + l) * DIM + (h << 3);
  #pragma unroll
  for (int nt = 0; nt < 4; ++nt)
    wp[nt] = W + (size_t)(n0 + (nt << 4) + l) * DIM + (h << 3);

  v8f acc[2][4];
  #pragma unroll
  for (int mt = 0; mt < 2; ++mt)
    #pragma unroll
    for (int nt = 0; nt < 4; ++nt) acc[mt][nt] = (v8f)V8F_ZERO;

  for (int kk = 0; kk < DIM; kk += 32) {
    BF16Frag fa[2], fb[4];
    #pragma unroll
    for (int mt = 0; mt < 2; ++mt) {
      fa[mt].q[0] = *(const u32x4*)(ap[mt] + kk);        // K = kk+8h .. +7
      fa[mt].q[1] = *(const u32x4*)(ap[mt] + kk + 16);   // K = kk+16+8h .. +7
    }
    #pragma unroll
    for (int nt = 0; nt < 4; ++nt) {
      fb[nt].q[0] = *(const u32x4*)(wp[nt] + kk);        // K = kk+8h .. +7
      fb[nt].q[1] = *(const u32x4*)(wp[nt] + kk + 16);   // K = kk+16+8h .. +7
    }
    #pragma unroll
    for (int mt = 0; mt < 2; ++mt)
      #pragma unroll
      for (int nt = 0; nt < 4; ++nt)
        acc[mt][nt] = wmma_bf16(fa[mt], fb[nt], acc[mt][nt]);
  }

  #pragma unroll
  for (int mt = 0; mt < 2; ++mt) {
    #pragma unroll
    for (int nt = 0; nt < 4; ++nt) {
      #pragma unroll
      for (int r = 0; r < 8; ++r) {
        const int m = m0 + (mt << 4) + (h << 3) + r;
        const int n = n0 + (nt << 4) + l;
        const float y = acc[mt][nt][r] + bias[n];
        const int b  = m >> 10;                 // / SEQ
        const int s  = m & (SEQ - 1);
        const int hd = n >> 6;
        const int dk = n & 63;
        if (mode == 3) {
          ((float*)Y0)[(size_t)m * DIM + n] = y;
        } else if (mode == 2) {
          ((unsigned short*)Y0)[((size_t)((b * NH + hd) * DK_ + dk)) * SEQ + s] = f2bf(y);
        } else {
          const size_t o = (((size_t)(b * NH + hd)) * SEQ + s) * DK_ + dk;
          ((unsigned short*)Y0)[o] = f2bf(y);
          if (mode == 1) Y1[o] = f2bf(sqrtf(fmaxf(y, 1e-24f)));
        }
      }
    }
  }
}

// ---------------------------------------------------------------------------
// Kernel 2: qsum = sum(qm^2)+sum(qc_raw), ksum = sum(km^2)+sum(kc_raw)
// ---------------------------------------------------------------------------
__global__ __launch_bounds__(256) void row_stats_kernel(
    const unsigned short* __restrict__ qm, const unsigned short* __restrict__ qc,
    const unsigned short* __restrict__ km, const unsigned short* __restrict__ kc,
    float* __restrict__ qsum, float* __restrict__ ksum)
{
  const int row = blockIdx.x * blockDim.x + threadIdx.x;
  if (row >= BH_ * SEQ) return;
  const size_t base = (size_t)row * DK_;
  float qs = 0.f, ks = 0.f;
  #pragma unroll 8
  for (int i = 0; i < DK_; ++i) {
    const float a = bf2f(qm[base + i]);
    qs += a * a + bf2f(qc[base + i]);
    const float b = bf2f(km[base + i]);
    ks += b * b + bf2f(kc[base + i]);
  }
  qsum[row] = qs;
  ksum[row] = ks;
}

// ---------------------------------------------------------------------------
// Kernel 3: scores[q,k] = (2*(qm.km + sqc.skc) - qs[q] - ks[k]) / 8
// One wave per 32x32 tile (4 accumulators x mean+cov); skips masked tiles.
// ---------------------------------------------------------------------------
__global__ __launch_bounds__(32) void scores_kernel(
    const unsigned short* __restrict__ qm, const unsigned short* __restrict__ sqc,
    const unsigned short* __restrict__ km, const unsigned short* __restrict__ skc,
    const float* __restrict__ qsum, const float* __restrict__ ksum,
    float* __restrict__ scores)
{
  const int qt = blockIdx.x >> 5;           // SEQ/32 = 32 tiles per side
  const int kt = blockIdx.x & 31;
  if (kt > qt) return;                      // fully masked tile, never read
  const int bh = blockIdx.y;
  const int q0 = qt << 5, k0 = kt << 5;

  const int lane = threadIdx.x;
  const int l = lane & 15, h = lane >> 4;

  const unsigned short* qmp[2];
  const unsigned short* qcp[2];
  const unsigned short* kmp[2];
  const unsigned short* kcp[2];
  #pragma unroll
  for (int t = 0; t < 2; ++t) {
    const size_t qrow = ((size_t)bh * SEQ + q0 + (t << 4) + l) * DK_;
    const size_t krow = ((size_t)bh * SEQ + k0 + (t << 4) + l) * DK_;
    qmp[t] = qm  + qrow + (h << 3);
    qcp[t] = sqc + qrow + (h << 3);
    kmp[t] = km  + krow + (h << 3);
    kcp[t] = skc + krow + (h << 3);
  }

  v8f acc[2][2];
  #pragma unroll
  for (int mt = 0; mt < 2; ++mt)
    #pragma unroll
    for (int nt = 0; nt < 2; ++nt) acc[mt][nt] = (v8f)V8F_ZERO;

  #pragma unroll
  for (int step = 0; step < 2; ++step) {
    const int ko = step << 5;
    BF16Frag am[2], ac[2], bm[2], bc[2];
    #pragma unroll
    for (int t = 0; t < 2; ++t) {
      am[t].q[0] = *(const u32x4*)(qmp[t] + ko);
      am[t].q[1] = *(const u32x4*)(qmp[t] + ko + 16);
      ac[t].q[0] = *(const u32x4*)(qcp[t] + ko);
      ac[t].q[1] = *(const u32x4*)(qcp[t] + ko + 16);
      bm[t].q[0] = *(const u32x4*)(kmp[t] + ko);
      bm[t].q[1] = *(const u32x4*)(kmp[t] + ko + 16);
      bc[t].q[0] = *(const u32x4*)(kcp[t] + ko);
      bc[t].q[1] = *(const u32x4*)(kcp[t] + ko + 16);
    }
    #pragma unroll
    for (int mt = 0; mt < 2; ++mt) {
      #pragma unroll
      for (int nt = 0; nt < 2; ++nt) {
        acc[mt][nt] = wmma_bf16(am[mt], bm[nt], acc[mt][nt]);
        acc[mt][nt] = wmma_bf16(ac[mt], bc[nt], acc[mt][nt]);
      }
    }
  }

  const float* qsp = qsum + (size_t)bh * SEQ;
  const float* ksp = ksum + (size_t)bh * SEQ;
  #pragma unroll
  for (int mt = 0; mt < 2; ++mt) {
    #pragma unroll
    for (int nt = 0; nt < 2; ++nt) {
      #pragma unroll
      for (int r = 0; r < 8; ++r) {
        const int q = q0 + (mt << 4) + (h << 3) + r;
        const int k = k0 + (nt << 4) + l;
        scores[((size_t)bh * SEQ + q) * SEQ + k] =
            (2.f * acc[mt][nt][r] - qsp[q] - ksp[k]) * 0.125f;
      }
    }
  }
}

// ---------------------------------------------------------------------------
// Kernel 4: fused masked softmax + cumsum + distance decay + second softmax.
// One 256-thread block per (bh, q) row; writes attention 'a' in place (fp32).
// ---------------------------------------------------------------------------
__global__ __launch_bounds__(256) void softmax_decay_kernel(
    float* __restrict__ scores, const float* __restrict__ gammas)
{
  const int bh = blockIdx.x >> 10;
  const int q  = blockIdx.x & (SEQ - 1);
  float* row = scores + ((size_t)bh * SEQ + q) * SEQ;
  const int tid = threadIdx.x;
  const int kbase = tid << 2;

  if (q == 0) {                       // zero_pad: first query row forced to 0
    fv4 z = {0.f, 0.f, 0.f, 0.f};
    *(fv4*)(row + kbase) = z;
    return;
  }

  __shared__ float red[256];
  __shared__ float sc[256];

  fv4 sv = *(const fv4*)(row + kbase);
  float s[4]; bool val[4];
  #pragma unroll
  for (int j = 0; j < 4; ++j) { s[j] = sv[j]; val[j] = (kbase + j) <= q; }

  float lm = -3.0e38f;
  #pragma unroll
  for (int j = 0; j < 4; ++j) if (val[j]) lm = fmaxf(lm, s[j]);
  red[tid] = lm; __syncthreads();
  for (int o = 128; o > 0; o >>= 1) {
    if (tid < o) red[tid] = fmaxf(red[tid], red[tid + o]);
    __syncthreads();
  }
  const float m1 = red[0]; __syncthreads();

  float e[4], ls = 0.f;
  #pragma unroll
  for (int j = 0; j < 4; ++j) { e[j] = val[j] ? __expf(s[j] - m1) : 0.f; ls += e[j]; }
  red[tid] = ls; __syncthreads();
  for (int o = 128; o > 0; o >>= 1) {
    if (tid < o) red[tid] += red[tid + o];
    __syncthreads();
  }
  const float Z = red[0]; __syncthreads();
  const float inv = 1.f / Z;

  float pre[4]; float run = 0.f;
  #pragma unroll
  for (int j = 0; j < 4; ++j) { run += e[j] * inv; pre[j] = run; }
  sc[tid] = run; __syncthreads();
  for (int o = 1; o < 256; o <<= 1) {
    float v = (tid >= o) ? sc[tid - o] : 0.f;
    __syncthreads();
    sc[tid] += v;
    __syncthreads();
  }
  const float excl  = sc[tid] - run;
  const float total = sc[255];

  const int   hd = bh & (NH - 1);
  const float g  = gammas[hd];
  const float sp = (g > 20.f) ? g : log1pf(__expf(g));  // softplus
  const float gamma = -sp;

  float s2[4]; float lm2 = -3.0e38f;
  #pragma unroll
  for (int j = 0; j < 4; ++j) {
    const float cum  = excl + pre[j];
    const float pos  = fabsf((float)((kbase + j) - q));
    const float dist = sqrtf(fmaxf((total - cum) * pos, 0.f));
    const float eff  = fminf(fmaxf(__expf(dist * gamma), 1e-5f), 1e5f);
    s2[j] = s[j] * eff;
    if (val[j]) lm2 = fmaxf(lm2, s2[j]);
  }
  red[tid] = lm2; __syncthreads();
  for (int o = 128; o > 0; o >>= 1) {
    if (tid < o) red[tid] = fmaxf(red[tid], red[tid + o]);
    __syncthreads();
  }
  const float m2 = red[0]; __syncthreads();

  float e2[4], ls2 = 0.f;
  #pragma unroll
  for (int j = 0; j < 4; ++j) { e2[j] = val[j] ? __expf(s2[j] - m2) : 0.f; ls2 += e2[j]; }
  red[tid] = ls2; __syncthreads();
  for (int o = 128; o > 0; o >>= 1) {
    if (tid < o) red[tid] += red[tid + o];
    __syncthreads();
  }
  const float inv2 = 1.f / red[0];

  fv4 outv;
  #pragma unroll
  for (int j = 0; j < 4; ++j) outv[j] = e2[j] * inv2;
  *(fv4*)(row + kbase) = outv;
}

// ---------------------------------------------------------------------------
// Kernel 5: om = a @ vm, oc = (a*a) @ vc.  One wave per 32-query tile:
// 16 accumulator chains, B fragments reused across both query sub-tiles.
// V stored transposed [bh][dk][s] so B fragments are contiguous b128 loads.
// Skips the all-zero K range beyond the causal diagonal. Emits bf16 outputs.
// ---------------------------------------------------------------------------
__global__ __launch_bounds__(32) void av_kernel(
    const float* __restrict__ att,
    const unsigned short* __restrict__ vmT, const unsigned short* __restrict__ vcT,
    unsigned short* __restrict__ om, unsigned short* __restrict__ oc)
{
  const int qt = blockIdx.x;                // SEQ/32 tiles
  const int bh = blockIdx.y;
  const int q0 = qt << 5;
  const int b  = bh >> 3;
  const int hd = bh & 7;

  const int lane = threadIdx.x;
  const int l = lane & 15, h = lane >> 4;

  v8f cm[2][4], cc[2][4];
  #pragma unroll
  for (int mt = 0; mt < 2; ++mt)
    #pragma unroll
    for (int nt = 0; nt < 4; ++nt) { cm[mt][nt] = (v8f)V8F_ZERO; cc[mt][nt] = (v8f)V8F_ZERO; }

  const int kmax = q0 + 32;                 // covers k <= q0+31 exactly
  const float* arow[2];
  #pragma unroll
  for (int mt = 0; mt < 2; ++mt)
    arow[mt] = att + ((size_t)bh * SEQ + q0 + (mt << 4) + l) * SEQ + (h << 3);

  for (int kk = 0; kk < kmax; kk += 32) {
    BF16Frag fa[2], fa2[2];
    #pragma unroll
    for (int mt = 0; mt < 2; ++mt) {
      #pragma unroll
      for (int j = 0; j < 8; ++j) {
        const float a0 = arow[mt][kk + j];        // K = kk+8h+j
        const float a1 = arow[mt][kk + 16 + j];   // K = kk+16+8h+j
        fa[mt].u[j]      = f2bf(a0);
        fa[mt].u[8 + j]  = f2bf(a1);
        fa2[mt].u[j]     = f2bf(a0 * a0);
        fa2[mt].u[8 + j] = f2bf(a1 * a1);
      }
    }
    #pragma unroll
    for (int nt = 0; nt < 4; ++nt) {
      const size_t vrow = ((size_t)(bh * DK_ + (nt << 4) + l)) * SEQ + kk + (h << 3);
      BF16Frag fb, fbc;
      fb.q[0]  = *(const u32x4*)(vmT + vrow);
      fb.q[1]  = *(const u32x4*)(vmT + vrow + 16);
      fbc.q[0] = *(const u32x4*)(vcT + vrow);
      fbc.q[1] = *(const u32x4*)(vcT + vrow + 16);
      #pragma unroll
      for (int mt = 0; mt < 2; ++mt) {
        cm[mt][nt] = wmma_bf16(fa[mt],  fb,  cm[mt][nt]);
        cc[mt][nt] = wmma_bf16(fa2[mt], fbc, cc[mt][nt]);
      }
    }
  }

  #pragma unroll
  for (int mt = 0; mt < 2; ++mt) {
    #pragma unroll
    for (int nt = 0; nt < 4; ++nt) {
      #pragma unroll
      for (int r = 0; r < 8; ++r) {
        const int m = q0 + (mt << 4) + (h << 3) + r;
        const int d = (hd << 6) + (nt << 4) + l;
        const size_t o = ((size_t)b * SEQ + m) * DIM + d;
        om[o] = f2bf(cm[mt][nt][r]);
        oc[o] = f2bf(cc[mt][nt][r]);
      }
    }
  }
}

// ---------------------------------------------------------------------------
extern "C" void kernel_launch(void* const* d_in, const int* in_sizes, int n_in,
                              void* d_out, int out_size, void* d_ws, size_t ws_size,
                              hipStream_t stream)
{
  (void)in_sizes; (void)n_in; (void)out_size; (void)ws_size;

  const float* xin[6] = {(const float*)d_in[0], (const float*)d_in[1],
                         (const float*)d_in[2], (const float*)d_in[3],
                         (const float*)d_in[4], (const float*)d_in[5]};
  const float* Wf[8]  = {(const float*)d_in[6],  (const float*)d_in[8],
                         (const float*)d_in[10], (const float*)d_in[12],
                         (const float*)d_in[14], (const float*)d_in[16],
                         (const float*)d_in[18], (const float*)d_in[20]};
  const float* bias[8] = {(const float*)d_in[7],  (const float*)d_in[9],
                          (const float*)d_in[11], (const float*)d_in[13],
                          (const float*)d_in[15], (const float*)d_in[17],
                          (const float*)d_in[19], (const float*)d_in[21]};
  const float* gammas = (const float*)d_in[22];
  // d_in[23] = mask: causal tril, applied analytically (k <= q).

  float* out = (float*)d_out;

  // ---- workspace carve-up (byte offsets; everything 16B aligned) ----
  char* ws = (char*)d_ws;
  const size_t XN = (size_t)MROWS * DIM;          // 4,194,304 elements
  const size_t WN = (size_t)DIM * DIM;            //   262,144 elements
  unsigned short* xbf[6];
  for (int i = 0; i < 6; ++i) { xbf[i] = (unsigned short*)ws; ws += XN * 2; }
  unsigned short* wbf[8];
  for (int i = 0; i < 8; ++i) { wbf[i] = (unsigned short*)ws; ws += WN * 2; }
  unsigned short* qm_bf  = (unsigned short*)ws; ws += XN * 2;
  unsigned short* qc_bf  = (unsigned short*)ws; ws += XN * 2;
  unsigned short* sqc_bf = (unsigned short*)ws; ws += XN * 2;
  unsigned short* km_bf  = (unsigned short*)ws; ws += XN * 2;
  unsigned short* kc_bf  = (unsigned short*)ws; ws += XN * 2;
  unsigned short* skc_bf = (unsigned short*)ws; ws += XN * 2;
  unsigned short* vmT_bf = (unsigned short*)ws; ws += XN * 2;
  unsigned short* vcT_bf = (unsigned short*)ws; ws += XN * 2;
  float* qsum = (float*)ws; ws += (size_t)BH_ * SEQ * 4;
  float* ksum = (float*)ws; ws += (size_t)BH_ * SEQ * 4;
  float* scores = (float*)ws; ws += (size_t)BH_ * SEQ * SEQ * 4;   // reused as 'a'
  unsigned short* om_bf = (unsigned short*)ws; ws += XN * 2;
  unsigned short* oc_bf = (unsigned short*)ws; ws += XN * 2;

  // Stage 0: bulk fp32 -> bf16 conversion of activations and weights.
  for (int i = 0; i < 6; ++i)
    cvt_bf16_kernel<<<dim3((unsigned)(XN / (256 * 8))), dim3(256), 0, stream>>>(
        xin[i], xbf[i], (int)XN);
  for (int i = 0; i < 8; ++i)
    cvt_bf16_kernel<<<dim3((unsigned)(WN / (256 * 8))), dim3(256), 0, stream>>>(
        Wf[i], wbf[i], (int)WN);

  const dim3 gemm_grid((MROWS / 32) * (DIM / 64));   // 2048 waves, 32x64 tiles
  const dim3 gemm_blk(32);

  // Stage A: six input projections.
  gemm_bf16_kernel<<<gemm_grid, gemm_blk, 0, stream>>>(xbf[0], wbf[0], bias[0], qm_bf, nullptr, 0);
  gemm_bf16_kernel<<<gemm_grid, gemm_blk, 0, stream>>>(xbf[1], wbf[1], bias[1], qc_bf, sqc_bf, 1);
  gemm_bf16_kernel<<<gemm_grid, gemm_blk, 0, stream>>>(xbf[2], wbf[2], bias[2], km_bf, nullptr, 0);
  gemm_bf16_kernel<<<gemm_grid, gemm_blk, 0, stream>>>(xbf[3], wbf[3], bias[3], kc_bf, skc_bf, 1);
  gemm_bf16_kernel<<<gemm_grid, gemm_blk, 0, stream>>>(xbf[4], wbf[4], bias[4], vmT_bf, nullptr, 2);
  gemm_bf16_kernel<<<gemm_grid, gemm_blk, 0, stream>>>(xbf[5], wbf[5], bias[5], vcT_bf, nullptr, 2);

  // Stage B: per-row rank-1 terms of mean_d + cov_d.
  row_stats_kernel<<<dim3((BH_ * SEQ) / 256), dim3(256), 0, stream>>>(
      qm_bf, qc_bf, km_bf, kc_bf, qsum, ksum);

  // Stage C: Wasserstein scores (fused mean + cov WMMA chains, 32x32/wave).
  scores_kernel<<<dim3((SEQ / 32) * (SEQ / 32), BH_), dim3(32), 0, stream>>>(
      qm_bf, sqc_bf, km_bf, skc_bf, qsum, ksum, scores);

  // Stage D: fused softmax / cumsum / distance-decay / softmax (in-place -> a).
  softmax_decay_kernel<<<dim3(BH_ * SEQ), dim3(256), 0, stream>>>(scores, gammas);

  // Stage E: om = a @ vm, oc = a^2 @ vc (bf16 outputs for final projections).
  av_kernel<<<dim3(SEQ / 32, BH_), dim3(32), 0, stream>>>(
      scores, vmT_bf, vcT_bf, om_bf, oc_bf);

  // Stage F: output projections straight into d_out (mean, then cov).
  gemm_bf16_kernel<<<gemm_grid, gemm_blk, 0, stream>>>(om_bf, wbf[6], bias[6], out, nullptr, 3);
  gemm_bf16_kernel<<<gemm_grid, gemm_blk, 0, stream>>>(
      oc_bf, wbf[7], bias[7], out + (size_t)MROWS * DIM, nullptr, 3);
}
